// LatentPredictorModel_86371792322915
// MI455X (gfx1250) — compile-verified, matrix operand order checked by reading, not run on hardware
//
#include <hip/hip_runtime.h>
#include <hip/hip_bf16.h>

// Problem constants
#define BB 256
#define TT 512
#define LAT 64
#define HH 512
#define RR 512
#define R3 1536

// Padded LDS strides (avoid 64-bank conflicts on A-fragment row reads)
#define SIN_LD 72     // bf16 elems per row for the [16][64] input tile
#define SH_LD  520    // bf16 elems per row for [16][512] activations / h
#define P_LD   1540   // f32 elems per row for the [16][1536] product buffer

typedef __attribute__((ext_vector_type(16))) __bf16   bf16x16;
typedef __attribute__((ext_vector_type(8)))  float    floatx8;
typedef __attribute__((ext_vector_type(4)))  unsigned uintx4;
typedef __attribute__((ext_vector_type(4)))  int      intx4;

#define AS1 __attribute__((address_space(1)))
#define AS3 __attribute__((address_space(3)))

#if __has_builtin(__builtin_amdgcn_global_load_async_to_lds_b128)
#define ASYNC_LDS 1
#else
#define ASYNC_LDS 0
#endif

__device__ __forceinline__ void wait_asynccnt0() {
#if __has_builtin(__builtin_amdgcn_s_wait_asynccnt)
    __builtin_amdgcn_s_wait_asynccnt(0);
#else
    asm volatile("s_wait_asynccnt 0x0" ::: "memory");
#endif
}

// float -> bf16: use native conversion (backend emits hardware cvt path)
__device__ __forceinline__ __bf16 f2bf(float f) { return (__bf16)f; }

__device__ __forceinline__ floatx8 wmma_bf16(bf16x16 a, bf16x16 b, floatx8 c) {
    // D = A(16x32 bf16) * B(32x16 bf16) + C(16x16 f32)
    return __builtin_amdgcn_wmma_f32_16x16x32_bf16(
        /*neg_a=*/false, a, /*neg_b=*/false, b,
        /*c_mod=*/(short)0, c, /*reuse_a=*/false, /*reuse_b=*/false);
}

// ---------------------------------------------------------------------------
// A-operand fragment from LDS (row-major bf16, row pitch ldaBytes).
// ISA layout (16-bit A 16x32): lane m=lane%16, half=lane/16,
//   VGPRs 0..3 hold K = 8*half + 0..7   (contiguous -> one ds b128)
//   VGPRs 4..7 hold K = 16 + 8*half + 0..7 (contiguous -> one ds b128)
// ---------------------------------------------------------------------------
__device__ __forceinline__ bf16x16 load_a_frag(const char* __restrict__ aB,
                                               int ldaBytes, int k0) {
    const int lane = threadIdx.x & 31;
    const int m    = lane & 15;
    const int half = lane >> 4;
    union { bf16x16 v; uintx4 q[2]; } f;
    const char* base = aB + m * ldaBytes + (k0 + 8 * half) * 2;
    f.q[0] = *(const uintx4*)(base);        // K = k0 + 8h .. +7
    f.q[1] = *(const uintx4*)(base + 32);   // K = k0 + 16 + 8h .. +7
    return f.v;
}

// B fragments are pre-swizzled: tile (nt,kt) stores 32 lanes x 16 bf16
// contiguously, k-major within a column tile -> two global_load_b128 per lane.
__device__ __forceinline__ bf16x16 load_b_frag(const __bf16* __restrict__ Bp,
                                               int ktiles, int nt, int kt) {
    const int lane = threadIdx.x & 31;
    const bf16x16* p = (const bf16x16*)(Bp +
        ((size_t)((size_t)nt * ktiles + kt) * 32 + lane) * 16);
    return *p;
}

// ---------------------------------------------------------------------------
// Pack fp32 row-major [K][N] weights into fragment-major bf16 B-operand form.
// ---------------------------------------------------------------------------
__global__ void pack_b_bf16(const float* __restrict__ src,
                            __bf16* __restrict__ dst, int K, int N) {
    int idx = blockIdx.x * blockDim.x + threadIdx.x;
    if (idx >= K * N) return;
    const int e    = idx & 15;
    const int lane = (idx >> 4) & 31;
    const int tile = idx >> 9;
    const int ktiles = K >> 5;
    const int kt = tile % ktiles;
    const int nt = tile / ktiles;
    const int half = lane >> 4;
    const int n    = lane & 15;
    const int v    = e >> 1;
    const int lohi = e & 1;
    const int kk   = (v < 4 ? 2 * v : 16 + 2 * (v - 4)) + lohi + 8 * half;
    const int k    = kt * 32 + kk;
    dst[idx] = f2bf(src[(size_t)k * N + nt * 16 + n]);
}

// ---------------------------------------------------------------------------
// Fused MLP + input projection:
//   x0 = relu(latent @ W0 + b0); x1 = relu(x0 @ W1 + b1); xi = x1 @ Wi + bi
// One WG (512 threads = 16 wave32) per 16-row tile of the flattened B*T axis.
// ---------------------------------------------------------------------------
__global__ void __launch_bounds__(512) mlp_proj_kernel(
    const float* __restrict__ latent,
    const float* __restrict__ b0, const float* __restrict__ b1,
    const float* __restrict__ bi,
    const __bf16* __restrict__ W0p, const __bf16* __restrict__ W1p,
    const __bf16* __restrict__ Wip,
    float* __restrict__ xi) {
    extern __shared__ char smem[];
    __bf16* sIn = (__bf16*)smem;                              // [16][SIN_LD]
    __bf16* sX0 = (__bf16*)(smem + 16 * SIN_LD * 2);          // [16][SH_LD]
    __bf16* sX1 = (__bf16*)(smem + 16 * SIN_LD * 2 + 16 * SH_LD * 2);

    const size_t row0 = (size_t)blockIdx.x * 16;
    const int tid  = threadIdx.x;
    const int wave = tid >> 5;
    const int lane = tid & 31;
    const int nlo  = lane & 15;
    const int mbase = (lane >> 4) * 8;

    for (int i = tid; i < 16 * LAT; i += 512) {
        int m = i >> 6, c = i & 63;
        sIn[m * SIN_LD + c] = f2bf(latent[row0 * LAT + i]);
    }
    __syncthreads();

    // Stage A: x0 = relu(in @ W0 + b0), K=64, 32 col tiles / 16 waves
    {
        const char* aB = (const char*)sIn;
#pragma unroll
        for (int t = 0; t < 2; ++t) {
            const int nt = wave * 2 + t;
            floatx8 acc = {};
#pragma unroll
            for (int kt = 0; kt < 2; ++kt)
                acc = wmma_bf16(load_a_frag(aB, SIN_LD * 2, kt * 32),
                                load_b_frag(W0p, 2, nt, kt), acc);
            const int n = nt * 16 + nlo;
            const float bias = b0[n];
#pragma unroll
            for (int r = 0; r < 8; ++r) {
                float v = acc[r] + bias;
                sX0[(mbase + r) * SH_LD + n] = f2bf(v > 0.f ? v : 0.f);
            }
        }
    }
    __syncthreads();

    // Stage B: x1 = relu(x0 @ W1 + b1), K=512
    {
        const char* aB = (const char*)sX0;
#pragma unroll
        for (int t = 0; t < 2; ++t) {
            const int nt = wave * 2 + t;
            floatx8 acc = {};
#pragma unroll 4
            for (int kt = 0; kt < 16; ++kt)
                acc = wmma_bf16(load_a_frag(aB, SH_LD * 2, kt * 32),
                                load_b_frag(W1p, 16, nt, kt), acc);
            const int n = nt * 16 + nlo;
            const float bias = b1[n];
#pragma unroll
            for (int r = 0; r < 8; ++r) {
                float v = acc[r] + bias;
                sX1[(mbase + r) * SH_LD + n] = f2bf(v > 0.f ? v : 0.f);
            }
        }
    }
    __syncthreads();

    // Stage C: xi = x1 @ Wi + bi, [16][1536]; 6 col tiles per wave, A reused
    {
        const char* aB = (const char*)sX1;
        floatx8 acc[6];
#pragma unroll
        for (int c = 0; c < 6; ++c) acc[c] = (floatx8){};
#pragma unroll 2
        for (int kt = 0; kt < 16; ++kt) {
            const bf16x16 a = load_a_frag(aB, SH_LD * 2, kt * 32);
#pragma unroll
            for (int c = 0; c < 6; ++c)
                acc[c] = wmma_bf16(a, load_b_frag(Wip, 16, wave * 6 + c, kt),
                                   acc[c]);
        }
#pragma unroll
        for (int c = 0; c < 6; ++c) {
            const int col = (wave * 6 + c) * 16 + nlo;
            const float bias = bi[col];
#pragma unroll
            for (int r = 0; r < 8; ++r)
                xi[(row0 + mbase + r) * R3 + col] = acc[c][r] + bias;
        }
    }
}

// ---------------------------------------------------------------------------
// GRU scan + output projection. One WG per 16-row batch tile (16 WGs total).
// Per step: async-stage xi[t] into LDS while 16 waves x 6 tiles x 16 k-steps
// of WMMA compute P = h @ [Whrz|Whn]; then elementwise sigmoid/tanh update.
// ---------------------------------------------------------------------------
__global__ void __launch_bounds__(512) gru_scan_kernel(
    const float* __restrict__ xi, const float* __restrict__ bhn,
    const float* __restrict__ bo,
    const __bf16* __restrict__ Whrzp, const __bf16* __restrict__ Whnp,
    const __bf16* __restrict__ Wop,
    float* __restrict__ out) {
    extern __shared__ char smem[];
    float*  hF  = (float*)smem;                               // [16][512]
    __bf16* hB  = (__bf16*)(smem + 32 * 1024);                // [16][SH_LD]
    float*  P   = (float*)(smem + 32 * 1024 + 16 * SH_LD * 2);// [16][P_LD]
    float*  sXi = (float*)(smem + 32 * 1024 + 16 * SH_LD * 2 +
                           16 * P_LD * 4);                    // [16][1536]

    const int bt   = blockIdx.x;  // batch tile 0..15
    const int tid  = threadIdx.x;
    const int wave = tid >> 5;
    const int lane = tid & 31;
    const int nlo  = lane & 15;
    const int mbase = (lane >> 4) * 8;

    for (int i = tid; i < 16 * SH_LD; i += 512) hB[i] = f2bf(0.f);
    for (int i = tid; i < 16 * RR; i += 512) hF[i] = 0.f;
    __syncthreads();

    const int erow = wave;       // elementwise: one wave per row
    const int ej0  = lane;

    for (int t = 0; t < TT; ++t) {
        // --- kick off async staging of xi[:, t, :] for this batch tile ---
        // (wave w copies row w: 1536 f32 = 384 b128, 12 per lane)
        const float* grow = xi + ((size_t)(bt * 16 + wave) * TT + t) * R3;
#if ASYNC_LDS
        {
            const char* g16 = (const char*)grow + lane * 16;
            char*       l16 = (char*)(sXi + wave * R3) + lane * 16;
#pragma unroll
            for (int i = 0; i < 12; ++i)
                __builtin_amdgcn_global_load_async_to_lds_b128(
                    (AS1 intx4*)(void*)(g16 + 512 * i),
                    (AS3 intx4*)(void*)(l16 + 512 * i), 0, 0);
        }
#endif

        // --- GEMM: P = h @ [Whrz | Whn] (overlaps with async copy) ---
        const char* aB = (const char*)hB;
        floatx8 acc[6];
#pragma unroll
        for (int c = 0; c < 6; ++c) acc[c] = (floatx8){};
#pragma unroll 2
        for (int kt = 0; kt < 16; ++kt) {
            const bf16x16 a = load_a_frag(aB, SH_LD * 2, kt * 32);
#pragma unroll
            for (int c = 0; c < 6; ++c) {
                const int ct = wave * 6 + c;           // lane-uniform
                const bf16x16 b = (ct < 64)
                    ? load_b_frag(Whrzp, 16, ct, kt)
                    : load_b_frag(Whnp, 16, ct - 64, kt);
                acc[c] = wmma_bf16(a, b, acc[c]);
            }
        }
#pragma unroll
        for (int c = 0; c < 6; ++c) {
            const int n = (wave * 6 + c) * 16 + nlo;
#pragma unroll
            for (int r = 0; r < 8; ++r) P[(mbase + r) * P_LD + n] = acc[c][r];
        }
#if ASYNC_LDS
        wait_asynccnt0();      // my async writes done; barrier orders the rest
#endif
        __syncthreads();

        // --- elementwise GRU update (one wave per row) ---
        {
            if (t + 1 < TT)  // pull next timestep of this row toward L2
                __builtin_prefetch(grow + R3, 0, 1);
#if ASYNC_LDS
            const float* xrow = sXi + erow * R3;
#else
            const float* xrow = xi + ((size_t)(bt * 16 + erow) * TT + t) * R3;
#endif
#pragma unroll 4
            for (int i = 0; i < 16; ++i) {
                const int j = ej0 + 32 * i;
                const float xr = xrow[j];
                const float xz = xrow[RR + j];
                const float xn = xrow[2 * RR + j];
                const float pr = P[erow * P_LD + j];
                const float pz = P[erow * P_LD + RR + j];
                const float pn = P[erow * P_LD + 2 * RR + j];
                const float r  = 1.f / (1.f + __expf(-(xr + pr)));
                const float z  = 1.f / (1.f + __expf(-(xz + pz)));
                const float a  = xn + r * (pn + bhn[j]);
                const float nn = 2.f / (1.f + __expf(-2.f * a)) - 1.f; // tanh
                const float ho = hF[erow * RR + j];
                const float hn = (1.f - z) * nn + z * ho;
                hF[erow * RR + j] = hn;
                hB[erow * SH_LD + j] = f2bf(hn);
            }
        }
        __syncthreads();
    }

    // --- out = h @ Wo + bo, [16][64]; waves 0..3 (wave-uniform branch) ---
    if (wave < 4) {
        const char* aB = (const char*)hB;
        floatx8 acc = {};
#pragma unroll 4
        for (int kt = 0; kt < 16; ++kt)
            acc = wmma_bf16(load_a_frag(aB, SH_LD * 2, kt * 32),
                            load_b_frag(Wop, 16, wave, kt), acc);
        const int n = wave * 16 + nlo;
        const float bias = bo[n];
#pragma unroll
        for (int r = 0; r < 8; ++r)
            out[(size_t)(bt * 16 + mbase + r) * LAT + n] = acc[r] + bias;
    }
}

// ---------------------------------------------------------------------------
extern "C" void kernel_launch(void* const* d_in, const int* in_sizes, int n_in,
                              void* d_out, int out_size, void* d_ws,
                              size_t ws_size, hipStream_t stream) {
    const float* latent = (const float*)d_in[0];
    const float* W0   = (const float*)d_in[1];
    const float* b0   = (const float*)d_in[2];
    const float* W1   = (const float*)d_in[3];
    const float* b1   = (const float*)d_in[4];
    const float* Wi   = (const float*)d_in[5];
    const float* bi   = (const float*)d_in[6];
    const float* Whrz = (const float*)d_in[7];
    const float* Whn  = (const float*)d_in[8];
    const float* bhn  = (const float*)d_in[9];
    const float* Wo   = (const float*)d_in[10];
    const float* bo   = (const float*)d_in[11];
    float* out = (float*)d_out;

    char* ws = (char*)d_ws;
    size_t off = 0;
    auto carve = [&](size_t bytes) {
        char* p = ws + off;
        off += (bytes + 255) & ~(size_t)255;
        return p;
    };
    __bf16* W0p   = (__bf16*)carve((size_t)LAT * HH * 2);
    __bf16* W1p   = (__bf16*)carve((size_t)HH * HH * 2);
    __bf16* Wip   = (__bf16*)carve((size_t)HH * R3 * 2);
    __bf16* Whrzp = (__bf16*)carve((size_t)RR * 2 * RR * 2);
    __bf16* Whnp  = (__bf16*)carve((size_t)RR * RR * 2);
    __bf16* Wop   = (__bf16*)carve((size_t)RR * LAT * 2);
    float*  xiBuf = (float*)carve((size_t)BB * TT * R3 * 4);

    // 1) pre-swizzle weights into WMMA B-operand fragment layout (bf16)
    auto packN = [&](const float* src, __bf16* dst, int K, int N) {
        int total = K * N;
        pack_b_bf16<<<(total + 255) / 256, 256, 0, stream>>>(src, dst, K, N);
    };
    packN(W0, W0p, LAT, HH);
    packN(W1, W1p, HH, HH);
    packN(Wi, Wip, HH, R3);
    packN(Whrz, Whrzp, RR, 2 * RR);
    packN(Whn, Whnp, RR, RR);
    packN(Wo, Wop, RR, LAT);

    // 2) fused MLP + input projection over all B*T rows
    {
        const size_t smemA = 16 * SIN_LD * 2 + 2 * (16 * SH_LD * 2); // ~35 KB
        dim3 grid((BB * TT) / 16);
        mlp_proj_kernel<<<grid, 512, smemA, stream>>>(latent, b0, b1, bi,
                                                      W0p, W1p, Wip, xiBuf);
    }

    // 3) GRU scan (serial over T inside each WG) + fused output projection
    {
        const size_t smemS = 32 * 1024 + 16 * SH_LD * 2 + 16 * P_LD * 4 +
                             16 * R3 * 4;                           // ~240 KB
        gru_scan_kernel<<<dim3(BB / 16), 512, smemS, stream>>>(
            xiBuf, bhn, bo, Whrzp, Whnp, Wop, out);
    }
}
